// Graph_mamba6__87119116632699
// MI455X (gfx1250) — matrix operand order repaired and checked by reading, not compiled, not run on hardware
//
#include <hip/hip_runtime.h>
#include <cstdint>
#include <cstddef>

// ---------------- model constants ----------------
#define BB      8
#define LL      2048
#define NN      (BB*LL)          // 16384 nodes per graph
#define EE      (NN*8)           // 131072 edges per graph
#define DIMM    512
#define DIM2    256
#define DINNER  512
#define DSTATE  16
#define DTRANK  16
#define NODE_DIM 64
#define WALK    20
#define PE_DIM  8

#define ACT_NONE 0
#define ACT_RELU 1
#define ACT_SILU 2
#define ACT_SOFTPLUS 3

typedef __attribute__((ext_vector_type(16))) __bf16 v16bf;
typedef __attribute__((ext_vector_type(8)))  float  v8f;
typedef unsigned short u16;

// ---------------- device helpers ----------------
__device__ __forceinline__ float siluf(float x){ return x / (1.f + __expf(-x)); }
__device__ __forceinline__ float softplusf(float x){ return (x > 20.f) ? x : log1pf(__expf(x)); }
__device__ __forceinline__ float geluf(float x){
  float x3 = x*x*x;
  return 0.5f*x*(1.f + tanhf(0.7978845608028654f*(x + 0.044715f*x3)));
}
__device__ __forceinline__ u16 f2bf(float f){
  union { float f; unsigned u; } c; c.f = f;
  unsigned u = c.u;
  unsigned r = u + 0x7FFFu + ((u >> 16) & 1u);   // round-to-nearest-even
  return (u16)(r >> 16);
}
__device__ __forceinline__ float actf(float v, int act){
  if (act == ACT_RELU) return fmaxf(v, 0.f);
  if (act == ACT_SILU) return siluf(v);
  if (act == ACT_SOFTPLUS) return softplusf(v);
  return v;
}

// ---------------- generic fills / adds ----------------
__global__ __launch_bounds__(256) void k_fill_u32(unsigned* p, size_t n){
  size_t i = (size_t)blockIdx.x*256 + threadIdx.x;
  if (i < n) p[i] = 0u;
}
__global__ __launch_bounds__(256) void k_add(const float* __restrict__ a,
                                             const float* __restrict__ b,
                                             float* __restrict__ c, size_t n){
  size_t i = (size_t)blockIdx.x*256 + threadIdx.x;
  if (i < n) c[i] = a[i] + b[i];
}

// ---------------- graph sort (replicates jnp.lexsort((deg, batch)), stable) ----------------
__global__ __launch_bounds__(256) void k_deg(const int* __restrict__ ei, int* __restrict__ deg){
  int k = blockIdx.x*256 + threadIdx.x;
  if (k >= EE) return;
  atomicAdd(&deg[ei[k]], 1);      // row 0 = sources
}
__global__ __launch_bounds__(256) void k_rank(const int* __restrict__ deg,
                                              const int* __restrict__ batch,
                                              int* __restrict__ perm, int* __restrict__ inv){
  int i = blockIdx.x*256 + threadIdx.x;
  if (i >= NN) return;
  int b = batch[i];
  int base = b * LL;
  int di = deg[i];
  int rank = 0;
  for (int j = base; j < base + LL; ++j){
    int dj = deg[j];
    rank += (dj < di) || (dj == di && j < i);   // stable
  }
  int r = base + rank;
  perm[r] = i;
  inv[i]  = r;
}

// h[r] = concat(x[perm[r]] @ node_w + node_b, pe[perm[r]] @ pe_w + pe_b)
__global__ __launch_bounds__(256) void k_node_embed(const float* __restrict__ x,
                                                    const float* __restrict__ pe,
                                                    const int* __restrict__ perm,
                                                    const float* __restrict__ node_w,
                                                    const float* __restrict__ node_b,
                                                    const float* __restrict__ pe_w,
                                                    const float* __restrict__ pe_b,
                                                    float* __restrict__ h){
  int id = blockIdx.x*256 + threadIdx.x;
  if (id >= NN*DIM2) return;
  int r = id >> 8;
  int d = id & 255;
  int src = perm[r];
  float acc;
  if (d < DIM2 - PE_DIM){
    acc = node_b[d];
    for (int j = 0; j < NODE_DIM; ++j)
      acc += x[(size_t)src*NODE_DIM + j] * node_w[j*(DIM2-PE_DIM) + d];
  } else {
    int d2 = d - (DIM2 - PE_DIM);
    acc = pe_b[d2];
    for (int j = 0; j < WALK; ++j)
      acc += pe[(size_t)src*WALK + j] * pe_w[j*PE_DIM + d2];
  }
  h[id] = acc;
}

// agg[dst] += relu(h[src] + ea*edge_w + edge_b)   (edge_w is (1,256) -> rank-1 edge embedding)
__global__ __launch_bounds__(256) void k_agg(const int* __restrict__ ei,
                                             const int* __restrict__ inv,
                                             const float* __restrict__ ea,
                                             const float* __restrict__ edge_w,
                                             const float* __restrict__ edge_b,
                                             const float* __restrict__ h,
                                             float* __restrict__ agg){
  size_t id = (size_t)blockIdx.x*256 + threadIdx.x;
  if (id >= (size_t)EE*DIM2) return;
  int d = (int)(id & 255);
  int k = (int)(id >> 8);
  int s   = inv[ei[k]];
  int dst = inv[ei[EE + k]];
  float m = h[(size_t)s*DIM2 + d] + ea[k]*edge_w[d] + edge_b[d];
  atomicAdd(&agg[(size_t)dst*DIM2 + d], fmaxf(m, 0.f));
}

// ---------------- WMMA GEMM (bf16 inputs, f32 accumulate) ----------------
// A packed fragment order (16-bit A 16x32, ISA 7.12.2):
//   idx = ((mt*KT + kt)*32 + lane)*16 + e ; m = mt*16 + (lane&15)
//   k   = kt*32 + (e&8?16:0) + (lane&16?8:0) + 2*((e>>1)&3) + (e&1)
__global__ __launch_bounds__(256) void k_apack(const float* __restrict__ A, u16* __restrict__ Ap,
                                               int M, int lda, int Kvalid, int KT){
  size_t idx = (size_t)blockIdx.x*256 + threadIdx.x;
  size_t total = (size_t)(M >> 4) * KT * 512;
  if (idx >= total) return;
  int e = (int)(idx & 15);
  int l = (int)((idx >> 4) & 31);
  size_t t = idx >> 9;
  int kt = (int)(t % KT);
  int mt = (int)(t / KT);
  int m = mt*16 + (l & 15);
  int k = kt*32 + ((e & 8) ? 16 : 0) + ((l & 16) ? 8 : 0) + (((e >> 1) & 3) << 1) + (e & 1);
  float v = (k < Kvalid) ? A[(size_t)m*lda + k] : 0.f;
  Ap[idx] = f2bf(v);
}
// B packed fragment order (same as A): n = nt*16 + (lane&15), k = kt*32 + (e&8?16:0) + (lane&16?8:0) + (e&7)
// NTpad tiles are emitted; columns >= Nvalid are zero-filled so the GEMM main loop is branch-free.
__global__ __launch_bounds__(256) void k_wpack(const float* __restrict__ W, u16* __restrict__ Wp,
                                               int NTpad, int ldw, int ncol0, int Kvalid,
                                               int Nvalid, int KT){
  size_t idx = (size_t)blockIdx.x*256 + threadIdx.x;
  size_t total = (size_t)NTpad * KT * 512;
  if (idx >= total) return;
  int e = (int)(idx & 15);
  int l = (int)((idx >> 4) & 31);
  size_t t = idx >> 9;
  int kt = (int)(t % KT);
  int nt = (int)(t / KT);
  int n = nt*16 + (l & 15);
  int k = kt*32 + ((e & 8) ? 16 : 0) + ((l & 16) ? 8 : 0) + (e & 7);
  float v = (k < Kvalid && n < Nvalid) ? W[(size_t)k*ldw + ncol0 + n] : 0.f;
  Wp[idx] = f2bf(v);
}

// C = act(A @ W + bias). One wave owns a 32x64 C tile:
//   2 A-frags + 4 B-frags -> 8 back-to-back WMMAs per kt (branch-free main loop).
__global__ __launch_bounds__(256) void k_gemm(const u16* __restrict__ Ap,
                                              const u16* __restrict__ Wp,
                                              const float* __restrict__ bias,
                                              float* __restrict__ C,
                                              int M, int Ncols, int KT, int NG, int act){
  const int lane = threadIdx.x & 31;
  const int wid  = (int)((blockIdx.x * 256 + threadIdx.x) >> 5);
  const int mt2  = wid / NG;            // 32-row group
  const int ng   = wid - mt2 * NG;      // 64-col group
  if (mt2 * 32 >= M) return;

  const size_t tstride = (size_t)KT * 512;     // u16 elems per 16-wide tile strip
  const u16* a0p = Ap + (size_t)(2*mt2    ) * tstride + (size_t)lane * 16;
  const u16* a1p = Ap + (size_t)(2*mt2 + 1) * tstride + (size_t)lane * 16;
  const u16* wp  = Wp + (size_t)(4*ng)     * tstride + (size_t)lane * 16;

  v8f acc[2][4];
  v8f vzero = {};
#pragma unroll
  for (int i = 0; i < 2; ++i)
#pragma unroll
    for (int j = 0; j < 4; ++j) acc[i][j] = vzero;

  for (int kt = 0; kt < KT; ++kt){
    const size_t ko = (size_t)kt * 512;
    v16bf a0 = *(const v16bf*)(a0p + ko);
    v16bf a1 = *(const v16bf*)(a1p + ko);
    v16bf b0 = *(const v16bf*)(wp + 0*tstride + ko);
    v16bf b1 = *(const v16bf*)(wp + 1*tstride + ko);
    v16bf b2 = *(const v16bf*)(wp + 2*tstride + ko);
    v16bf b3 = *(const v16bf*)(wp + 3*tstride + ko);
    if (kt + 1 < KT){
      __builtin_prefetch(a0p + ko + 512, 0, 1);
      __builtin_prefetch(a1p + ko + 512, 0, 1);
    }
    acc[0][0] = __builtin_amdgcn_wmma_f32_16x16x32_bf16(false, a0, false, b0, (short)0, acc[0][0], false, false);
    acc[0][1] = __builtin_amdgcn_wmma_f32_16x16x32_bf16(false, a0, false, b1, (short)0, acc[0][1], false, false);
    acc[0][2] = __builtin_amdgcn_wmma_f32_16x16x32_bf16(false, a0, false, b2, (short)0, acc[0][2], false, false);
    acc[0][3] = __builtin_amdgcn_wmma_f32_16x16x32_bf16(false, a0, false, b3, (short)0, acc[0][3], false, false);
    acc[1][0] = __builtin_amdgcn_wmma_f32_16x16x32_bf16(false, a1, false, b0, (short)0, acc[1][0], false, false);
    acc[1][1] = __builtin_amdgcn_wmma_f32_16x16x32_bf16(false, a1, false, b1, (short)0, acc[1][1], false, false);
    acc[1][2] = __builtin_amdgcn_wmma_f32_16x16x32_bf16(false, a1, false, b2, (short)0, acc[1][2], false, false);
    acc[1][3] = __builtin_amdgcn_wmma_f32_16x16x32_bf16(false, a1, false, b3, (short)0, acc[1][3], false, false);
  }

  // epilogue: D layout (ISA 7.12.2): VGPR r -> M = r (lanes 0-15) / r+8 (lanes 16-31), N = lane&15
  const int nl   = lane & 15;
  const int moff = (lane & 16) ? 8 : 0;
#pragma unroll
  for (int i = 0; i < 2; ++i){
    const int rbase = mt2*32 + i*16 + moff;
#pragma unroll
    for (int j = 0; j < 4; ++j){
      int ncol = (4*ng + j)*16 + nl;
      if (ncol >= Ncols) continue;          // only the zero-padded tail tile hits this
      float bv = bias ? bias[ncol] : 0.f;
#pragma unroll
      for (int r = 0; r < 8; ++r){
        C[(size_t)(rbase + r) * Ncols + ncol] = actf(acc[i][j][r] + bv, act);
      }
    }
  }
}

// ---------------- LayerNorm (LDS reduction): out = LN(a (+ b)) [* g + beta] ----------------
__global__ __launch_bounds__(256) void k_ln(const float* __restrict__ a,
                                            const float* __restrict__ b,
                                            const float* __restrict__ g,
                                            const float* __restrict__ beta,
                                            float* __restrict__ out, int D){
  __shared__ float ssum[256];
  __shared__ float ssq[256];
  const int row = blockIdx.x;
  const int t = threadIdx.x;
  const size_t base = (size_t)row * D;
  float s = 0.f, q = 0.f;
  for (int j = t; j < D; j += 256){
    float v = a[base + j] + (b ? b[base + j] : 0.f);
    s += v; q += v*v;
  }
  ssum[t] = s; ssq[t] = q;
  __syncthreads();
  for (int off = 128; off > 0; off >>= 1){
    if (t < off){ ssum[t] += ssum[t + off]; ssq[t] += ssq[t + off]; }
    __syncthreads();
  }
  float mean = ssum[0] / D;
  float var  = ssq[0] / D - mean*mean;
  float rstd = rsqrtf(var + 1e-5f);
  for (int j = t; j < D; j += 256){
    float v = a[base + j] + (b ? b[base + j] : 0.f);
    float y = (v - mean) * rstd;
    if (g) y = y * g[j] + beta[j];
    out[base + j] = y;
  }
}

// ---------------- mamba pieces ----------------
// xc = silu(depthwise causal conv4(xi) + conv_b)
__global__ __launch_bounds__(256) void k_conv(const float* __restrict__ xi,
                                              const float* __restrict__ conv_w,
                                              const float* __restrict__ conv_b,
                                              float* __restrict__ xc){
  size_t id = (size_t)blockIdx.x*256 + threadIdx.x;
  if (id >= (size_t)NN*DINNER) return;
  int c = (int)(id & (DINNER-1));
  size_t r = id >> 9;
  int l = (int)(r & (LL-1));
  float acc = conv_b[c];
#pragma unroll
  for (int k = 0; k < 4; ++k){
    int lk = l - 3 + k;
    if (lk >= 0) acc += conv_w[k*DINNER + c] * xi[(r - 3 + k)*DINNER + c];
  }
  xc[id] = siluf(acc);
}

// selective-scan: one thread per (b,d), 16 states in registers, sequential over L
__global__ __launch_bounds__(256) void k_scan(const float* __restrict__ dt,
                                              const float* __restrict__ xc,
                                              const float* __restrict__ dbc,
                                              const float* __restrict__ A_log,
                                              float* __restrict__ ys){
  int id = blockIdx.x*256 + threadIdx.x;
  if (id >= BB*DINNER) return;
  int b = id / DINNER, d = id % DINNER;
  float A[DSTATE], s[DSTATE];
#pragma unroll
  for (int j = 0; j < DSTATE; ++j){ A[j] = -__expf(A_log[d*DSTATE + j]); s[j] = 0.f; }
  for (int l = 0; l < LL; ++l){
    size_t r = (size_t)b*LL + l;
    float dtv = dt[r*DINNER + d];
    float xv  = xc[r*DINNER + d];
    const float* Brow = dbc + r*48 + DTRANK;
    const float* Crow = dbc + r*48 + DTRANK + DSTATE;
    float y = 0.f;
#pragma unroll
    for (int j = 0; j < DSTATE; ++j){
      s[j] = s[j]*__expf(dtv*A[j]) + dtv*xv*Brow[j];
      y += s[j]*Crow[j];
    }
    ys[r*DINNER + d] = y;
  }
}

// xc <- (ys + xc*D) * silu(z)
__global__ __launch_bounds__(256) void k_gate(const float* __restrict__ ys,
                                              float* __restrict__ xc,
                                              const float* __restrict__ z,
                                              const float* __restrict__ Dp){
  size_t id = (size_t)blockIdx.x*256 + threadIdx.x;
  if (id >= (size_t)NN*DINNER) return;
  int d = (int)(id & (DINNER-1));
  float y = ys[id] + xc[id]*Dp[d];
  xc[id] = y * siluf(z[id]);
}

// ---------------- pooling / head ----------------
__global__ __launch_bounds__(256) void k_pool(const float* __restrict__ h, float* __restrict__ pooled){
  int id = blockIdx.x*256 + threadIdx.x;
  if (id >= BB*DIM2) return;
  int b = id / DIM2, d = id % DIM2;
  float s = 0.f;
  for (int l = 0; l < LL; ++l) s += h[((size_t)b*LL + l)*DIM2 + d];
  pooled[id] = s / (float)LL;
}

__global__ __launch_bounds__(256) void k_embed_x(const int* __restrict__ x_categ,
                                                 const float* __restrict__ x_numer,
                                                 const float* __restrict__ cat_embed,
                                                 const float* __restrict__ num_w,
                                                 const float* __restrict__ num_b,
                                                 float* __restrict__ xout){
  int id = blockIdx.x*256 + threadIdx.x;
  if (id >= BB*DIMM) return;
  int b = id / DIMM, d = id % DIMM;
  const int offs[4] = {2, 7, 17, 23};   // cumsum([NUM_SPECIAL]+CATS[:-1])
  float s = 0.f;
  for (int t = 0; t < 4; ++t){
    int tok = x_categ[b*4 + t] + offs[t];
    s += cat_embed[(size_t)tok*DIMM + d];
  }
  for (int j = 0; j < 8; ++j)
    s += x_numer[b*8 + j]*num_w[j*DIMM + d] + num_b[j*DIMM + d];
  xout[id] = s / 12.f;
}

__global__ __launch_bounds__(256) void k_concat2(const float* __restrict__ a,
                                                 const float* __restrict__ b,
                                                 float* __restrict__ c){
  int id = blockIdx.x*256 + threadIdx.x;
  if (id >= BB*2*DIM2) return;
  int bb = id / (2*DIM2), d = id % (2*DIM2);
  c[id] = (d < DIM2) ? a[bb*DIM2 + d] : b[bb*DIM2 + d - DIM2];
}

// tiny dense: C[b,n] = sum_k A[b,k]W[k,n] (+bias) (+add), arbitrary strides (B=8 rows only)
__global__ __launch_bounds__(256) void k_matvec(const float* __restrict__ A,
                                                const float* __restrict__ W,
                                                const float* __restrict__ bias,
                                                const float* __restrict__ add,
                                                float* __restrict__ C,
                                                int Mr, int Nc, int K,
                                                int lda, int ldw, int ldadd, int ldc){
  int id = blockIdx.x*256 + threadIdx.x;
  if (id >= Mr*Nc) return;
  int b = id / Nc, n = id % Nc;
  float acc = bias ? bias[n] : 0.f;
  for (int k = 0; k < K; ++k) acc += A[(size_t)b*lda + k] * W[(size_t)k*ldw + n];
  if (add) acc += add[(size_t)b*ldadd + n];
  C[(size_t)b*ldc + n] = acc;
}

__global__ __launch_bounds__(256) void k_glu(const float* __restrict__ h1, float* __restrict__ ag){
  int id = blockIdx.x*256 + threadIdx.x;
  if (id >= BB*2048) return;
  int b = id / 2048, j = id % 2048;
  float a = h1[b*4096 + j];
  float g = h1[b*4096 + 2048 + j];
  ag[id] = a * geluf(g);
}

// ---------------- host side ----------------
struct MambaP { const float *A_log,*D,*conv_b,*conv_w,*dt_bias,*dt_proj,*in_proj,*out_proj,*x_proj; };
struct LayerP { const float *gine_b1,*gine_b2,*gine_w1,*gine_w2; MambaP mamba;
                const float *mlp_b1,*mlp_b2,*mlp_w1,*mlp_w2; };
struct GnnP   { const float *edge_b,*edge_w; LayerP layers[2];
                const float *node_b,*node_w,*pe_b,*pe_w; };

static inline int divup_sz(size_t a, int b){ return (int)((a + (size_t)b - 1) / (size_t)b); }

extern "C" void kernel_launch(void* const* d_in, const int* in_sizes, int n_in,
                              void* d_out, int out_size, void* d_ws, size_t ws_size,
                              hipStream_t stream){
  (void)in_sizes; (void)n_in; (void)out_size; (void)ws_size;

  // ---- parse params: jax pytree leaf order (dict keys sorted, lists in order) ----
  int idx = 0;
  auto nf = [&]{ return (const float*)d_in[idx++]; };
  auto parse_mamba = [&](MambaP& m){
    m.A_log=nf(); m.D=nf(); m.conv_b=nf(); m.conv_w=nf(); m.dt_bias=nf();
    m.dt_proj=nf(); m.in_proj=nf(); m.out_proj=nf(); m.x_proj=nf();
  };
  auto parse_layer = [&](LayerP& l){
    l.gine_b1=nf(); l.gine_b2=nf(); l.gine_w1=nf(); l.gine_w2=nf();
    parse_mamba(l.mamba);
    l.mlp_b1=nf(); l.mlp_b2=nf(); l.mlp_w1=nf(); l.mlp_w2=nf();
  };
  auto parse_gnn = [&](GnnP& g){
    g.edge_b=nf(); g.edge_w=nf();
    parse_layer(g.layers[0]); parse_layer(g.layers[1]);
    g.node_b=nf(); g.node_w=nf(); g.pe_b=nf(); g.pe_w=nf();
  };

  const float* ca_k = nf();  const float* ca_o = nf();  const float* ca_ob = nf();
  const float* ca_q = nf();  const float* ca_v = nf();  const float* cat_embed = nf();
  const float* ff_b1 = nf(); const float* ff_b2 = nf(); const float* ff_w1 = nf();
  const float* ff_w2 = nf(); const float* ln_b = nf();  const float* ln_g = nf();
  GnnP mriP; parse_gnn(mriP);
  const float* num_b = nf(); const float* num_w = nf();
  const float* out_b = nf(); const float* out_w = nf();
  GnnP petP; parse_gnn(petP);
  (void)ca_k; (void)ca_q;   // seq lens are 1 -> softmax == 1 -> q,k drop out exactly

  const int*   x_categ  = (const int*)d_in[idx++];
  const float* x_numer  = nf();
  const float* mri_x    = nf();  const float* mri_pe = nf();
  const int*   mri_ei   = (const int*)d_in[idx++];
  const float* mri_ea   = nf();
  const int*   mri_bat  = (const int*)d_in[idx++];
  const float* pet_x    = nf();  const float* pet_pe = nf();
  const int*   pet_ei   = (const int*)d_in[idx++];
  const float* pet_ea   = nf();
  const int*   pet_bat  = (const int*)d_in[idx++];

  // ---- workspace arena (~230 MB) ----
  char* base = (char*)d_ws;
  size_t off = 0;
  auto take = [&](size_t bytes)->void*{
    void* r = base + off; off = (off + bytes + 255) & ~(size_t)255; return r;
  };
  int*   deg  = (int*)take((size_t)NN*4);
  int*   perm = (int*)take((size_t)NN*4);
  int*   inv  = (int*)take((size_t)NN*4);
  float* h    = (float*)take((size_t)NN*DIM2*4);
  float* agg  = (float*)take((size_t)NN*DIM2*4);
  float* hl   = (float*)take((size_t)NN*DIM2*4);
  float* ha   = (float*)take((size_t)NN*DIM2*4);
  float* t0   = (float*)take((size_t)NN*DINNER*4);
  float* t1   = (float*)take((size_t)NN*DINNER*4);
  float* t2   = (float*)take((size_t)NN*DINNER*4);
  float* t3   = (float*)take((size_t)NN*DINNER*4);
  float* dbc  = (float*)take((size_t)NN*48*4);
  u16*   Apk  = (u16*)take((size_t)NN*DINNER*2);   // max packed-A (K=512)
  u16*   Wpk  = (u16*)take((size_t)1<<20);
  float* pool_mri = (float*)take((size_t)BB*DIM2*4);
  float* pool_pet = (float*)take((size_t)BB*DIM2*4);
  float* xvec = (float*)take((size_t)BB*DIMM*4);
  float* cond = (float*)take((size_t)BB*2*DIM2*4);
  float* v1   = (float*)take((size_t)BB*DIMM*4);
  float* fx   = (float*)take((size_t)BB*1024*4);
  float* fx2  = (float*)take((size_t)BB*1024*4);
  float* lnfx = (float*)take((size_t)BB*1024*4);
  float* h1   = (float*)take((size_t)BB*4096*4);
  float* ag   = (float*)take((size_t)BB*2048*4);

  // ---- GEMM helpers ----
  auto apack = [&](const float* A, int M, int lda, int Kvalid){
    int KT = (Kvalid + 31)/32;
    size_t total = (size_t)(M/16)*KT*512;
    k_apack<<<divup_sz(total,256),256,0,stream>>>(A, Apk, M, lda, Kvalid, KT);
  };
  auto wpack = [&](const float* W, int Ncols, int ldw, int ncol0, int Kvalid){
    int KT = (Kvalid + 31)/32;
    int NT = (Ncols + 15)/16;
    int NG = (NT + 3)/4;
    int NTpad = NG*4;                          // zero-pad ragged N so GEMM loop is branch-free
    size_t total = (size_t)NTpad*KT*512;
    k_wpack<<<divup_sz(total,256),256,0,stream>>>(W, Wpk, NTpad, ldw, ncol0, Kvalid, Ncols, KT);
  };
  auto gemm = [&](const float* bias, float* C, int M, int Ncols, int K, int act){
    int KT = (K + 31)/32;
    int NT = (Ncols + 15)/16;
    int NG = (NT + 3)/4;
    size_t waves = (size_t)(M/32)*NG;          // one wave = 32x64 C tile
    int blocks = divup_sz(waves*32, 256);
    k_gemm<<<blocks,256,0,stream>>>(Apk, Wpk, bias, C, M, Ncols, KT, NG, act);
  };

  // ---- one GraphGPS encoder ----
  auto run_gnn = [&](const GnnP& G, const float* gx, const float* gpe, const int* gei,
                     const float* gea, const int* gbat, float* pooled){
    k_fill_u32<<<divup_sz(NN,256),256,0,stream>>>((unsigned*)deg, NN);
    k_deg <<<divup_sz(EE,256),256,0,stream>>>(gei, deg);
    k_rank<<<divup_sz(NN,256),256,0,stream>>>(deg, gbat, perm, inv);
    k_node_embed<<<divup_sz((size_t)NN*DIM2,256),256,0,stream>>>(gx, gpe, perm,
        G.node_w, G.node_b, G.pe_w, G.pe_b, h);

    for (int li = 0; li < 2; ++li){
      const LayerP& L = G.layers[li];
      // --- GINE branch ---
      k_fill_u32<<<divup_sz((size_t)NN*DIM2,256),256,0,stream>>>((unsigned*)agg, (size_t)NN*DIM2);
      k_agg<<<divup_sz((size_t)EE*DIM2,256),256,0,stream>>>(gei, inv, gea, G.edge_w, G.edge_b, h, agg);
      k_add<<<divup_sz((size_t)NN*DIM2,256),256,0,stream>>>(h, agg, t2, (size_t)NN*DIM2);
      apack(t2, NN, DIM2, DIM2);  wpack(L.gine_w1, DIM2, DIM2, 0, DIM2);
      gemm(L.gine_b1, t0, NN, DIM2, DIM2, ACT_RELU);
      apack(t0, NN, DIM2, DIM2);  wpack(L.gine_w2, DIM2, DIM2, 0, DIM2);
      gemm(L.gine_b2, t2, NN, DIM2, DIM2, ACT_NONE);
      k_ln<<<NN,256,0,stream>>>(t2, h, nullptr, nullptr, hl, DIM2);
      // --- mamba branch ---
      apack(h, NN, DIM2, DIM2);
      wpack(L.mamba.in_proj, DINNER, 2*DINNER, 0,      DIM2);
      gemm(nullptr, t0, NN, DINNER, DIM2, ACT_NONE);        // xi
      wpack(L.mamba.in_proj, DINNER, 2*DINNER, DINNER, DIM2);
      gemm(nullptr, t3, NN, DINNER, DIM2, ACT_NONE);        // z
      k_conv<<<divup_sz((size_t)NN*DINNER,256),256,0,stream>>>(t0, L.mamba.conv_w, L.mamba.conv_b, t1);
      apack(t1, NN, DINNER, DINNER);  wpack(L.mamba.x_proj, 48, 48, 0, DINNER);
      gemm(nullptr, dbc, NN, 48, DINNER, ACT_NONE);
      apack(dbc, NN, 48, DTRANK);     wpack(L.mamba.dt_proj, DINNER, DINNER, 0, DTRANK);
      gemm(L.mamba.dt_bias, t2, NN, DINNER, DTRANK, ACT_SOFTPLUS);   // dt
      k_scan<<<divup_sz(BB*DINNER,256),256,0,stream>>>(t2, t1, dbc, L.mamba.A_log, t0);
      k_gate<<<divup_sz((size_t)NN*DINNER,256),256,0,stream>>>(t0, t1, t3, L.mamba.D);
      apack(t1, NN, DINNER, DINNER);  wpack(L.mamba.out_proj, DIM2, DIM2, 0, DINNER);
      gemm(nullptr, t2, NN, DIM2, DINNER, ACT_NONE);
      k_ln<<<NN,256,0,stream>>>(t2, h, nullptr, nullptr, ha, DIM2);
      // --- combine + MLP ---
      k_add<<<divup_sz((size_t)NN*DIM2,256),256,0,stream>>>(hl, ha, t2, (size_t)NN*DIM2);
      apack(t2, NN, DIM2, DIM2);  wpack(L.mlp_w1, DINNER, DINNER, 0, DIM2);
      gemm(L.mlp_b1, t0, NN, DINNER, DIM2, ACT_RELU);
      apack(t0, NN, DINNER, DINNER);  wpack(L.mlp_w2, DIM2, DIM2, 0, DINNER);
      gemm(L.mlp_b2, t1, NN, DIM2, DINNER, ACT_NONE);
      k_ln<<<NN,256,0,stream>>>(t2, t1, nullptr, nullptr, h, DIM2);
    }
    k_pool<<<divup_sz(BB*DIM2,256),256,0,stream>>>(h, pooled);
  };

  run_gnn(mriP, mri_x, mri_pe, mri_ei, mri_ea, mri_bat, pool_mri);
  run_gnn(petP, pet_x, pet_pe, pet_ei, pet_ea, pet_bat, pool_pet);

  // ---- fusion head (B=8, all tiny) ----
  k_embed_x<<<divup_sz(BB*DIMM,256),256,0,stream>>>(x_categ, x_numer, cat_embed, num_w, num_b, xvec);
  k_concat2<<<divup_sz(BB*2*DIM2,256),256,0,stream>>>(pool_mri, pool_pet, cond);
  // res1 = (cond@ca_v)@ca_o + ca_ob + x   -> fx[:, :512]
  k_matvec<<<divup_sz(BB*DIMM,256),256,0,stream>>>(cond, ca_v, nullptr, nullptr, v1,
      BB, DIMM, DIMM, DIMM, DIMM, 0, DIMM);
  k_matvec<<<divup_sz(BB*DIMM,256),256,0,stream>>>(v1, ca_o, ca_ob, xvec, fx,
      BB, DIMM, DIMM, DIMM, DIMM, DIMM, 1024);
  // res2 = (x@ca_v)@ca_o + ca_ob + cond   -> fx[:, 512:]
  k_matvec<<<divup_sz(BB*DIMM,256),256,0,stream>>>(xvec, ca_v, nullptr, nullptr, v1,
      BB, DIMM, DIMM, DIMM, DIMM, 0, DIMM);
  k_matvec<<<divup_sz(BB*DIMM,256),256,0,stream>>>(v1, ca_o, ca_ob, cond, fx + DIMM,
      BB, DIMM, DIMM, DIMM, DIMM, DIMM, 1024);
  // fx -> LN -> ff_w1 -> GLU(gelu) -> ff_w2 (+fx) -> LN(g,b) -> out_w
  k_ln<<<BB,256,0,stream>>>(fx, nullptr, nullptr, nullptr, lnfx, 1024);
  k_matvec<<<divup_sz(BB*4096,256),256,0,stream>>>(lnfx, ff_w1, ff_b1, nullptr, h1,
      BB, 4096, 1024, 1024, 4096, 0, 4096);
  k_glu<<<divup_sz(BB*2048,256),256,0,stream>>>(h1, ag);
  k_matvec<<<divup_sz(BB*1024,256),256,0,stream>>>(ag, ff_w2, ff_b2, fx, fx2,
      BB, 1024, 2048, 2048, 1024, 1024, 1024);
  k_ln<<<BB,256,0,stream>>>(fx2, nullptr, ln_g, ln_b, lnfx, 1024);
  k_matvec<<<divup_sz(BB,256),256,0,stream>>>(lnfx, out_w, out_b, nullptr, (float*)d_out,
      BB, 1, 1024, 1024, 1, 0, 1);
}
